// OffsetPredictor_2216203124765
// MI455X (gfx1250) — matrix operand order repaired and checked by reading, not compiled, not run on hardware
//
#include <hip/hip_runtime.h>

// ---------------- problem constants (from reference) ----------------
#define B_      8
#define C_      256
#define L_      4096
#define PATCH_  16
#define STRIDE_ 8
#define HID_    64
#define OUT_    2
#define P_      511          // (L - PATCH)/STRIDE + 1

typedef __attribute__((ext_vector_type(16))) _Float16 v16h;
typedef __attribute__((ext_vector_type(8)))  float    v8f;
typedef int v4i __attribute__((vector_size(4 * sizeof(int))));
typedef __attribute__((address_space(1))) v4i* gptr_v4i;
typedef __attribute__((address_space(3))) v4i* lptr_v4i;

// ---------------- gfx1250 feature guards ----------------
#if defined(__has_builtin)
#  if __has_builtin(__builtin_amdgcn_global_load_async_to_lds_b128)
#    define HAVE_ASYNC_LDS 1
#  endif
#  if __has_builtin(__builtin_amdgcn_s_wait_asynccnt)
#    define WAIT_ASYNC0() __builtin_amdgcn_s_wait_asynccnt(0)
#  else
#    define WAIT_ASYNC0() asm volatile("s_wait_asynccnt 0" ::: "memory")
#  endif
#  if __has_builtin(__builtin_amdgcn_tanhf)
#    define TANHF(x) __builtin_amdgcn_tanhf(x)
#  else
#    define TANHF(x) tanhf(x)
#  endif
#else
#  define WAIT_ASYNC0() asm volatile("s_wait_asynccnt 0" ::: "memory")
#  define TANHF(x) tanhf(x)
#endif

__global__ __launch_bounds__(256) void patch_mlp_gfx1250(
    const float* __restrict__ x,   // (B, C, L)
    const float* __restrict__ W1,  // (HID, PATCH)
    const float* __restrict__ b1,  // (HID)
    const float* __restrict__ W2,  // (OUT, HID)
    const float* __restrict__ b2,  // (OUT)
    float* __restrict__ out)       // (B, P, C, OUT)
{
    __shared__ float xrow[L_ + 8];           // +8 zero pad for masked tail patch

    const int bc   = blockIdx.x;             // one (b,c) pair per workgroup
    const int tid  = threadIdx.x;
    const int lane = tid & 31;
    const int wid  = tid >> 5;               // wave id 0..7 (wave32)
    const int pl   = lane & 15;              // patch slot / hid row within tile
    const int hi   = lane >> 4;              // half-wave: 0 -> hid 0..7, 1 -> hid 8..15

    const float* xg = x + (size_t)bc * L_;

    // ---- stage the 16KB x-row into LDS (async LDS-DMA path when available) ----
#if HAVE_ASYNC_LDS
    #pragma unroll
    for (int i = 0; i < 4; ++i) {
        const int idx = i * 256 + tid;       // float4 index, 1024 total
        __builtin_amdgcn_global_load_async_to_lds_b128(
            (gptr_v4i)(xg + idx * 4),
            (lptr_v4i)(&xrow[idx * 4]),
            0, 0);
    }
#else
    #pragma unroll
    for (int i = 0; i < 4; ++i) {
        const int idx = i * 256 + tid;
        ((float4*)xrow)[idx] = ((const float4*)xg)[idx];
    }
#endif
    if (tid < 8) xrow[L_ + tid] = 0.0f;      // zero pad (patch 511 is masked anyway)

    // ---- loop-invariant per-wave fragments (independent of LDS contents) ----
    // A = W1 tile T (16 hid rows x K=16, zero-padded to K=32), A-layout:
    //   lane<16 : K 0..7 in elems 0..7 ; lane>=16 : K 8..15 in elems 0..7
    v16h afrag[4];
    v8f  biasf[4], w20f[4], w21f[4];
    #pragma unroll
    for (int T = 0; T < 4; ++T) {
        const float* wp = W1 + ((16 * T + pl) * PATCH_) + 8 * hi;
        v16h a;
        #pragma unroll
        for (int k = 0; k < 8; ++k)  a[k] = (_Float16)wp[k];
        #pragma unroll
        for (int k = 8; k < 16; ++k) a[k] = (_Float16)0.0f;   // K 16..31 pad
        afrag[T] = a;

        const float* bp  = b1 + 16 * T + 8 * hi;              // hh = 16T + 8*hi + r
        const float* w0p = W2 + 0 * HID_ + 16 * T + 8 * hi;
        const float* w1p = W2 + 1 * HID_ + 16 * T + 8 * hi;
        v8f bb, v0, v1;
        #pragma unroll
        for (int k = 0; k < 8; ++k) { bb[k] = bp[k]; v0[k] = w0p[k]; v1[k] = w1p[k]; }
        biasf[T] = bb; w20f[T] = v0; w21f[T] = v1;
    }
    const float b2v = b2[hi];                // lane's output channel j = hi

#if HAVE_ASYNC_LDS
    WAIT_ASYNC0();
#endif
    __syncthreads();

    // ---- 32 M-tiles of 16 patches; 4 per wave ----
    #pragma unroll
    for (int it = 0; it < 4; ++it) {
        const int p0 = (wid * 4 + it) * 16;

        // B = patches^T (K=16 s-values x N=16 patches, padded to K=32).
        // B-layout: lanes 0..15 hold K 0..7, lanes 16..31 hold K 8..15 in elems 0..7; elems 8..15 = 0 (pad).
        v16h bf;
        #pragma unroll
        for (int k = 0; k < 16; ++k) bf[k] = (_Float16)0.0f;
        {
            const float4* xp = (const float4*)&xrow[(p0 + pl) * STRIDE_ + 8 * hi];
            #pragma unroll
            for (int q = 0; q < 2; ++q) {
                float4 v = xp[q];
                bf[4 * q + 0] = (_Float16)v.x;
                bf[4 * q + 1] = (_Float16)v.y;
                bf[4 * q + 2] = (_Float16)v.z;
                bf[4 * q + 3] = (_Float16)v.w;
            }
        }

        // h^T tile: D[m=hid_local, n=patch] = W1_tile x patches^T + b1  (C = bias)
        v8f acc[4];
        #pragma unroll
        for (int T = 0; T < 4; ++T)
            acc[T] = __builtin_amdgcn_wmma_f32_16x16x32_f16(
                false, afrag[T], false, bf, (short)0, biasf[T], false, false);

        // GELU (tanh form, native v_tanh_f32) + lane-local 64x2 projection.
        float o0 = 0.0f, o1 = 0.0f;
        #pragma unroll
        for (int T = 0; T < 4; ++T) {
            #pragma unroll
            for (int e = 0; e < 8; ++e) {
                float h  = acc[T][e];
                float h2 = h * h;
                float u  = h * fmaf(0.0356774081f, h2, 0.7978845608f);
                float g  = 0.5f * h * (1.0f + TANHF(u));
                o0 = fmaf(g, w20f[T][e], o0);
                o1 = fmaf(g, w21f[T][e], o1);
            }
        }
        // partner half-wave holds the other 32 hidden values of the same patch
        o0 += __shfl_xor(o0, 16);
        o1 += __shfl_xor(o1, 16);

        const int p = p0 + pl;
        if (p < P_) {
            const float ov = (hi ? o1 : o0) + b2v;
            const int b = bc >> 8;           // / C_ (C_ == 256)
            const int c = bc & (C_ - 1);
            out[(((size_t)b * P_ + p) * C_ + c) * OUT_ + hi] = ov;
        }
    }
}

extern "C" void kernel_launch(void* const* d_in, const int* /*in_sizes*/, int /*n_in*/,
                              void* d_out, int /*out_size*/, void* /*d_ws*/, size_t /*ws_size*/,
                              hipStream_t stream) {
    const float* x  = (const float*)d_in[0];
    const float* W1 = (const float*)d_in[1];
    const float* b1 = (const float*)d_in[2];
    const float* W2 = (const float*)d_in[3];
    const float* b2 = (const float*)d_in[4];
    float* out = (float*)d_out;

    dim3 grid(B_ * C_);                      // 2048 workgroups, one x-row each
    dim3 block(256);                         // 8 wave32 waves
    hipLaunchKernelGGL(patch_mlp_gfx1250, grid, block, 0, stream,
                       x, W1, b1, W2, b2, out);
}
